// DSAATSPModel_10161892622812
// MI455X (gfx1250) — compile-verified, matrix operand order checked by reading, not run on hardware
//
#include <hip/hip_runtime.h>
#include <hip/hip_bf16.h>
#include <math.h>
#include <stdint.h>

// Problem constants (match reference)
#define B_ 8
#define N_ 1024
#define E_ 1024
#define H_ 16
#define D_ 64

typedef __attribute__((ext_vector_type(16))) _Float16     v16h;
typedef __attribute__((ext_vector_type(8)))  float        v8f;
typedef __attribute__((ext_vector_type(4)))  unsigned int u32x4;  // POD b128
typedef __attribute__((ext_vector_type(4)))  float        f32x4;

// Explicit LDS (addrspace 3) pointer types: guarantees ds_load/ds_store
// lowering for the WMMA operand fetches (never the FLAT path, which would
// charge both DScnt and LOADcnt and run aperture checks).
typedef __attribute__((address_space(3))) _Float16 ldsh;
typedef __attribute__((address_space(3))) u32x4    ldsv4;

__device__ __forceinline__ ldsh* to_lds(void* p) {
  // generic->LDS: low 32 bits of a generic LDS address are the wave-relative
  // LDS byte offset (flat-aperture rule), which is exactly the AS3 address.
  return (ldsh*)(uintptr_t)((unsigned long long)(uintptr_t)p & 0xffffffffull);
}

union HF16 {
  v16h     v;
  u32x4    q[2];
  _Float16 h[16];
};

__device__ __forceinline__ v8f wmma_f16(v16h a, v16h b, v8f c) {
  // D = A(16x32 f16) * B(32x16 f16) + C(16x16 f32)
  return __builtin_amdgcn_wmma_f32_16x16x32_f16(
      /*neg_a=*/false, a, /*neg_b=*/false, b,
      /*c_mod=*/(short)0, c, /*reuse_a=*/false, /*reuse_b=*/false);
}

// Load one fragment: CDNA5 16-bit A layout, lane L holds row (L&15); v16h
// elements 0..7 are K = base+0..7, elements 8..15 are K = base+16..23 with
// base = (L>=16 ? 8 : 0).  Caller passes p already offset by
// row*stride + (lane>>4)*8, so this is two contiguous ds_load_b128.
__device__ __forceinline__ v16h ld_frag(const ldsh* p) {
  HF16 u;
  u.q[0] = *(const ldsv4*)(p);
  u.q[1] = *(const ldsv4*)(p + 16);
  return u.v;
}

// Asynchronous global -> LDS copy of 16 bytes per lane (ASYNCcnt-tracked).
__device__ __forceinline__ void async_ld16(const _Float16* g, ldsh* l) {
  unsigned lds_addr = (unsigned)(uintptr_t)l;           // AS3 ptr == LDS offset
  unsigned long long ga = (unsigned long long)(uintptr_t)g;
  asm volatile("global_load_async_to_lds_b128 %0, %1, off"
               :: "v"(lds_addr), "v"(ga) : "memory");
}
__device__ __forceinline__ void wait_async0() {
  asm volatile("s_wait_asynccnt 0x0" ::: "memory");
}

// ---------------------------------------------------------------------------
// Generic 128x128 block GEMM mainloop, f16 inputs, f32 accumulate.
// A: [M,K] row-major.  B given as Bst[outCol][k] (row = output column) so
// both LDS tiles share one layout and one fragment loader.  256 threads =
// 8 waves in a 2(M) x 4(N) grid; each wave owns 64x32 = 4x2 WMMA tiles.
// LDS tiles are double-buffered and filled with async global->LDS DMA so the
// copy of tile k+1 overlaps the WMMAs of tile k.
// ---------------------------------------------------------------------------
#define LDT 40   // padded LDS K-stride in halfs (32 + 8) -> conflict-light b128
#define TILE_H (128 * LDT)

struct Acc8 { v8f c[4][2]; };

__device__ __forceinline__ void zero_acc(Acc8& acc) {
#pragma unroll
  for (int mi = 0; mi < 4; ++mi)
#pragma unroll
    for (int ni = 0; ni < 2; ++ni)
#pragma unroll
      for (int r = 0; r < 8; ++r) acc.c[mi][ni][r] = 0.0f;
}

__device__ __forceinline__ void stage_tile_async(
    const _Float16* Ag, const _Float16* Bg, int k0, int lda, int ldb,
    ldsh* As, ldsh* Bs, int ldr, int ldc) {
  const _Float16* ga = Ag + (long)ldr * lda + k0 + ldc;
  const _Float16* gb = Bg + (long)ldr * ldb + k0 + ldc;
  ldsh* la = As + ldr * LDT + ldc;
  ldsh* lb = Bs + ldr * LDT + ldc;
  async_ld16(ga,     la);
  async_ld16(ga + 8, la + 8);
  async_ld16(gb,     lb);
  async_ld16(gb + 8, lb + 8);
}

__device__ __forceinline__ void gemm128_mainloop(
    const _Float16* __restrict__ Ag, const _Float16* __restrict__ Bg,
    int K, int lda, int ldb, ldsh* lds /* 4*TILE_H halfs */, Acc8& acc) {
  const int tid  = threadIdx.x;
  const int lane = tid & 31;
  const int wave = tid >> 5;
  const int wm = wave & 1;          // 64-row slab
  const int wn = wave >> 1;         // 32-col slab
  const int lr = lane & 15;
  const int lh = lane >> 4;
  const int ldr = tid >> 1;         // 0..127 tile row to stage
  const int ldc = (tid & 1) * 16;   // first/second 16 halfs of the K slice

  stage_tile_async(Ag, Bg, 0, lda, ldb, lds, lds + TILE_H, ldr, ldc);
  int cur = 0;

  for (int k0 = 0; k0 < K; k0 += 32) {
    wait_async0();        // our slice of tile k0 has landed
    __syncthreads();      // everyone's slice landed; prev buffer fully read
    ldsh* As = cur ? lds + 2 * TILE_H : lds;
    ldsh* Bs = cur ? lds + 3 * TILE_H : lds + TILE_H;
    if (k0 + 32 < K)
      stage_tile_async(Ag, Bg, k0 + 32, lda, ldb,
                       cur ? lds : lds + 2 * TILE_H,
                       cur ? lds + TILE_H : lds + 3 * TILE_H, ldr, ldc);

    v16h af[4], bf[2];
#pragma unroll
    for (int mi = 0; mi < 4; ++mi)
      af[mi] = ld_frag(As + (wm * 64 + mi * 16 + lr) * LDT + lh * 8);
#pragma unroll
    for (int ni = 0; ni < 2; ++ni)
      bf[ni] = ld_frag(Bs + (wn * 32 + ni * 16 + lr) * LDT + lh * 8);
#pragma unroll
    for (int mi = 0; mi < 4; ++mi)
#pragma unroll
      for (int ni = 0; ni < 2; ++ni)
        acc.c[mi][ni] = wmma_f16(af[mi], bf[ni], acc.c[mi][ni]);
    cur ^= 1;
  }
}

// ---------------------------------------------------------------------------
// fp32 -> f16 conversion, 8 elements/thread, b128 in / b128 out
// ---------------------------------------------------------------------------
__global__ __launch_bounds__(256) void k_cvt_f16(const float* __restrict__ src,
                                                 _Float16* __restrict__ dst, long n) {
  long i = ((long)blockIdx.x * blockDim.x + threadIdx.x) * 8;
  if (i >= n) return;
  f32x4 a = *(const f32x4*)(src + i);
  f32x4 b = *(const f32x4*)(src + i + 4);
  union { u32x4 q; _Float16 h[8]; } o;
  o.h[0] = (_Float16)a.x; o.h[1] = (_Float16)a.y;
  o.h[2] = (_Float16)a.z; o.h[3] = (_Float16)a.w;
  o.h[4] = (_Float16)b.x; o.h[5] = (_Float16)b.y;
  o.h[6] = (_Float16)b.z; o.h[7] = (_Float16)b.w;
  *(u32x4*)(dst + i) = o.q;
}

// ---------------------------------------------------------------------------
// Time-embedding MLP: t[B] -> t_emb[B,E].  Tiny, scalar.
// ---------------------------------------------------------------------------
__global__ __launch_bounds__(256) void k_temb(
    const float* __restrict__ t, const float* __restrict__ tW1,
    const float* __restrict__ tb1, const float* __restrict__ tW2,
    const float* __restrict__ tb2, float* __restrict__ t_emb) {
  __shared__ float emb0[E_];
  __shared__ float hid[E_ / 2];
  const int b = blockIdx.x;
  const float tv = t[b];
  const float cexp = -logf(10000.0f) / (float)(E_ / 2);
  for (int e = threadIdx.x; e < E_; e += blockDim.x) {
    int i = (e < E_ / 2) ? e : (e - E_ / 2);
    float arg = tv * __expf(cexp * (float)i);
    emb0[e] = (e < E_ / 2) ? cosf(arg) : sinf(arg);
  }
  __syncthreads();
  for (int j = threadIdx.x; j < E_ / 2; j += blockDim.x) {
    float a = tb1[j];
    for (int k2 = 0; k2 < E_; ++k2) a += emb0[k2] * tW1[j * E_ + k2];
    hid[j] = fmaxf(a, 0.0f);
  }
  __syncthreads();
  for (int e = threadIdx.x; e < E_; e += blockDim.x) {
    float a = tb2[e];
    for (int k2 = 0; k2 < E_ / 2; ++k2) a += hid[k2] * tW2[e * (E_ / 2) + k2];
    t_emb[b * E_ + e] = a;
  }
}

// ---------------------------------------------------------------------------
// QKV projection: dst[b,h,n,d] = sum_e ej[b,n,e] * W[h*64+d, e]
// ---------------------------------------------------------------------------
__global__ __launch_bounds__(256) void k_qkv_gemm(
    const _Float16* __restrict__ ej16, const _Float16* __restrict__ w16,
    _Float16* __restrict__ dst) {
  __shared__ __align__(16) _Float16 smem[4 * TILE_H];
  Acc8 acc; zero_acc(acc);
  const _Float16* Ag = ej16 + (long)blockIdx.x * 128 * E_;
  const _Float16* Bg = w16  + (long)blockIdx.y * 128 * E_;   // W row = out col
  gemm128_mainloop(Ag, Bg, E_, E_, E_, to_lds(smem), acc);

  const int tid = threadIdx.x, lane = tid & 31, wave = tid >> 5;
  const int wm = wave & 1, wn = wave >> 1, lr = lane & 15, lh = lane >> 4;
#pragma unroll
  for (int mi = 0; mi < 4; ++mi)
#pragma unroll
    for (int ni = 0; ni < 2; ++ni)
#pragma unroll
      for (int r = 0; r < 8; ++r) {
        int row = blockIdx.x * 128 + wm * 64 + mi * 16 + r + lh * 8;
        int col = blockIdx.y * 128 + wn * 32 + ni * 16 + lr;
        int b = row >> 10, n = row & (N_ - 1);
        int h = col >> 6,  d = col & (D_ - 1);
        dst[(((long)(b * H_ + h) * N_) + n) * D_ + d] = (_Float16)acc.c[mi][ni][r];
      }
}

// ---------------------------------------------------------------------------
// Flash attention: one block = one (b,h), 128 Q rows (16 per wave).
// Scores never touch HBM; online softmax in VGPRs.  K/V 32-key tiles are
// double-buffered in LDS via async global->LDS DMA.
// ---------------------------------------------------------------------------
#define KVT (32 * 72)

__global__ __launch_bounds__(256) void k_attn(
    const _Float16* __restrict__ q16, const _Float16* __restrict__ k16,
    const _Float16* __restrict__ v16, _Float16* __restrict__ oc16) {
  __shared__ __align__(16) _Float16 smem[4 * KVT + 8 * 16 * LDT];
  ldsh* Kt0 = to_lds(smem);
  ldsh* Kt1 = Kt0 + KVT;
  ldsh* Vt0 = Kt0 + 2 * KVT;
  ldsh* Vt1 = Kt0 + 3 * KVT;
  ldsh* Pt  = Kt0 + 4 * KVT;

  const int tid = threadIdx.x, lane = tid & 31, wave = tid >> 5;
  const int lr = lane & 15, lh = lane >> 4;
  const int h = blockIdx.y, b = blockIdx.z;
  const long base = (long)(b * H_ + h) * N_ * D_;
  const int q0 = blockIdx.x * 128 + wave * 16;

  // Q fragments, kept in VGPRs for the whole kernel (A layout, ld_frag k order: 8h+s, 16+8h+(s-8))
  v16h qa[2];
  {
    const _Float16* qp = q16 + base + (long)(q0 + lr) * D_ + lh * 8;
    HF16 u;
    u.q[0] = *(const u32x4*)(qp);      u.q[1] = *(const u32x4*)(qp + 16); qa[0] = u.v;
    u.q[0] = *(const u32x4*)(qp + 32); u.q[1] = *(const u32x4*)(qp + 48); qa[1] = u.v;
  }

  v8f O[4];
#pragma unroll
  for (int di = 0; di < 4; ++di)
#pragma unroll
    for (int r = 0; r < 8; ++r) O[di][r] = 0.0f;
  float m_s[8], l_s[8];
#pragma unroll
  for (int r = 0; r < 8; ++r) { m_s[r] = -1e30f; l_s[r] = 0.0f; }

  ldsh* Pw = Pt + wave * 16 * LDT;
  const int srow = tid >> 3, sc8 = (tid & 7) * 8;  // staging slice

  // prologue: stage keys 0..31 into buffer 0
  async_ld16(k16 + base + (long)srow * D_ + sc8, Kt0 + srow * 72 + sc8);
  async_ld16(v16 + base + (long)srow * D_ + sc8, Vt0 + srow * 72 + sc8);
  int cur = 0;

  for (int j0 = 0; j0 < N_; j0 += 32) {
    wait_async0();
    __syncthreads();
    ldsh* Kt = cur ? Kt1 : Kt0;
    ldsh* Vt = cur ? Vt1 : Vt0;
    if (j0 + 32 < N_) {
      async_ld16(k16 + base + (long)(j0 + 32 + srow) * D_ + sc8,
                 (cur ? Kt0 : Kt1) + srow * 72 + sc8);
      async_ld16(v16 + base + (long)(j0 + 32 + srow) * D_ + sc8,
                 (cur ? Vt0 : Vt1) + srow * 72 + sc8);
    }

    // S = Q (16x64) x K^T : two 16x16 score tiles (keys j0..+15, j0+16..+31)
    v8f S0 = {}, S1 = {};
#pragma unroll
    for (int kk = 0; kk < 2; ++kk) {
      v16h kb0 = ld_frag(Kt + (lr)      * 72 + kk * 32 + lh * 8);
      v16h kb1 = ld_frag(Kt + (16 + lr) * 72 + kk * 32 + lh * 8);
      S0 = wmma_f16(qa[kk], kb0, S0);
      S1 = wmma_f16(qa[kk], kb1, S1);
    }

    // online softmax: row r of the C fragment lives in element r, spread
    // across a 16-lane half -> width-16 xor-shuffle reductions
    float alpha[8];
#pragma unroll
    for (int r = 0; r < 8; ++r) {
      float s0 = S0[r] * 0.125f, s1 = S1[r] * 0.125f;  // 1/SQRT_D
      float mx = fmaxf(s0, s1);
      mx = fmaxf(mx, __shfl_xor(mx, 1, 16));
      mx = fmaxf(mx, __shfl_xor(mx, 2, 16));
      mx = fmaxf(mx, __shfl_xor(mx, 4, 16));
      mx = fmaxf(mx, __shfl_xor(mx, 8, 16));
      float mn = fmaxf(m_s[r], mx);
      float al = __expf(m_s[r] - mn);
      float p0 = __expf(s0 - mn), p1 = __expf(s1 - mn);
      float rs = p0 + p1;
      rs += __shfl_xor(rs, 1, 16);
      rs += __shfl_xor(rs, 2, 16);
      rs += __shfl_xor(rs, 4, 16);
      rs += __shfl_xor(rs, 8, 16);
      l_s[r] = l_s[r] * al + rs;
      m_s[r] = mn;
      alpha[r] = al;
      // C-layout -> A-layout via per-wave LDS patch
      Pw[(r + lh * 8) * LDT + lr]      = (_Float16)p0;
      Pw[(r + lh * 8) * LDT + 16 + lr] = (_Float16)p1;
    }
#pragma unroll
    for (int di = 0; di < 4; ++di)
#pragma unroll
      for (int r = 0; r < 8; ++r) O[di][r] *= alpha[r];

    // per-wave LDS region: only need our own DS ops complete, not a barrier
    asm volatile("s_wait_dscnt 0x0" ::: "memory");

    v16h pa = ld_frag(Pw + lr * LDT + lh * 8);

    // O += P (16x32) x V (32 keys x 64 d), 4 d-tiles
#pragma unroll
    for (int di = 0; di < 4; ++di) {
      HF16 u;
#pragma unroll
      for (int e = 0; e < 16; ++e) {
        int key = (e < 8 ? e : e + 8) + lh * 8;   // B-fragment K mapping
        u.h[e] = Vt[key * 72 + di * 16 + lr];
      }
      O[di] = wmma_f16(pa, u.v, O[di]);
    }
    cur ^= 1;
  }

  float inv[8];
#pragma unroll
  for (int r = 0; r < 8; ++r) inv[r] = 1.0f / l_s[r];
#pragma unroll
  for (int di = 0; di < 4; ++di)
#pragma unroll
    for (int r = 0; r < 8; ++r) {
      int row = q0 + r + lh * 8;
      int d = di * 16 + lr;
      oc16[((long)b * N_ + row) * (H_ * D_) + h * D_ + d] =
          (_Float16)(O[di][r] * inv[r]);
    }
}

// ---------------------------------------------------------------------------
// mh = out_concat @ Wc^T + bc + t_emb  (f16 out for the score GEMM)
// ---------------------------------------------------------------------------
__global__ __launch_bounds__(256) void k_mh_gemm(
    const _Float16* __restrict__ oc16, const _Float16* __restrict__ wc16,
    const float* __restrict__ bc, const float* __restrict__ t_emb,
    _Float16* __restrict__ mh16) {
  __shared__ __align__(16) _Float16 smem[4 * TILE_H];
  Acc8 acc; zero_acc(acc);
  const _Float16* Ag = oc16 + (long)blockIdx.x * 128 * E_;
  const _Float16* Bg = wc16 + (long)blockIdx.y * 128 * E_;  // Wc row = out col
  gemm128_mainloop(Ag, Bg, E_, E_, E_, to_lds(smem), acc);

  const int tid = threadIdx.x, lane = tid & 31, wave = tid >> 5;
  const int wm = wave & 1, wn = wave >> 1, lr = lane & 15, lh = lane >> 4;
#pragma unroll
  for (int mi = 0; mi < 4; ++mi)
#pragma unroll
    for (int ni = 0; ni < 2; ++ni)
#pragma unroll
      for (int r = 0; r < 8; ++r) {
        int row = blockIdx.x * 128 + wm * 64 + mi * 16 + r + lh * 8;
        int col = blockIdx.y * 128 + wn * 32 + ni * 16 + lr;
        int b = row >> 10;
        float v = acc.c[mi][ni][r] + bc[col] + t_emb[b * E_ + col];
        mh16[(long)row * E_ + col] = (_Float16)v;
      }
}

// ---------------------------------------------------------------------------
// score = mh @ ej^T / 32, fused: 10*tanh, 1x1 conv(2ch) with xt, softmax(ch)
// ---------------------------------------------------------------------------
__global__ __launch_bounds__(256) void k_score_conv(
    const _Float16* __restrict__ mh16, const _Float16* __restrict__ ej16,
    const float* __restrict__ xt, const float* __restrict__ conv_w,
    const float* __restrict__ conv_b, float* __restrict__ out) {
  __shared__ __align__(16) _Float16 smem[4 * TILE_H];
  Acc8 acc; zero_acc(acc);
  const int b = blockIdx.z;
  const _Float16* Ag = mh16 + ((long)b * N_ + blockIdx.x * 128) * E_;
  const _Float16* Bg = ej16 + ((long)b * N_ + blockIdx.y * 128) * E_; // ej row = out col
  gemm128_mainloop(Ag, Bg, E_, E_, E_, to_lds(smem), acc);

  const float w00 = conv_w[0], w01 = conv_w[1], w10 = conv_w[2], w11 = conv_w[3];
  const float cb0 = conv_b[0], cb1 = conv_b[1];
  const int tid = threadIdx.x, lane = tid & 31, wave = tid >> 5;
  const int wm = wave & 1, wn = wave >> 1, lr = lane & 15, lh = lane >> 4;
#pragma unroll
  for (int mi = 0; mi < 4; ++mi)
#pragma unroll
    for (int ni = 0; ni < 2; ++ni)
#pragma unroll
      for (int r = 0; r < 8; ++r) {
        int n = blockIdx.x * 128 + wm * 64 + mi * 16 + r + lh * 8;
        int m = blockIdx.y * 128 + wn * 32 + ni * 16 + lr;
        float sc  = acc.c[mi][ni][r] * (1.0f / 32.0f);      // / SQRT_E
        float scl = 10.0f * tanhf(sc);                      // CLIP * tanh
        float x   = xt[((long)b * N_ + n) * N_ + m];
        float c0  = w00 * scl + w01 * x + cb0;
        float c1  = w10 * scl + w11 * x + cb1;
        float mx  = fmaxf(c0, c1);
        float e0  = __expf(c0 - mx), e1 = __expf(c1 - mx);
        float inv = 1.0f / (e0 + e1);
        long  o   = (((long)b * N_ + n) * N_ + m) * 2;
        out[o]     = e0 * inv;
        out[o + 1] = e1 * inv;
      }
}

// ---------------------------------------------------------------------------
extern "C" void kernel_launch(void* const* d_in, const int* in_sizes, int n_in,
                              void* d_out, int out_size, void* d_ws, size_t ws_size,
                              hipStream_t stream) {
  (void)in_sizes; (void)n_in; (void)out_size; (void)ws_size;
  const float* t   = (const float*)d_in[0];
  const float* ej  = (const float*)d_in[1];
  const float* xt  = (const float*)d_in[2];
  const float* Wq  = (const float*)d_in[3];
  const float* Wk  = (const float*)d_in[4];
  const float* Wv  = (const float*)d_in[5];
  const float* Wc  = (const float*)d_in[6];
  const float* bc  = (const float*)d_in[7];
  const float* cw  = (const float*)d_in[8];
  const float* cb  = (const float*)d_in[9];
  const float* tW1 = (const float*)d_in[10];
  const float* tb1 = (const float*)d_in[11];
  const float* tW2 = (const float*)d_in[12];
  const float* tb2 = (const float*)d_in[13];
  float* out = (float*)d_out;

  char* ws = (char*)d_ws;
  const long MB = 1024L * 1024L;
  _Float16* ej16 = (_Float16*)(ws + 0   * MB);  // 16 MB
  _Float16* q16  = (_Float16*)(ws + 16  * MB);  // 16 MB
  _Float16* k16  = (_Float16*)(ws + 32  * MB);  // 16 MB
  _Float16* v16  = (_Float16*)(ws + 48  * MB);  // 16 MB
  _Float16* oc16 = (_Float16*)(ws + 64  * MB);  // 16 MB
  _Float16* mh16 = (_Float16*)(ws + 80  * MB);  // 16 MB
  _Float16* wq16 = (_Float16*)(ws + 96  * MB);  // 2 MB
  _Float16* wk16 = (_Float16*)(ws + 98  * MB);  // 2 MB
  _Float16* wv16 = (_Float16*)(ws + 100 * MB);  // 2 MB
  _Float16* wc16 = (_Float16*)(ws + 102 * MB);  // 2 MB
  float*    temb = (float*)  (ws + 104 * MB);   // 32 KB

  const long nEJ = (long)B_ * N_ * E_;
  const long nW  = (long)E_ * E_;
  k_cvt_f16<<<(int)(nEJ / 8 / 256), 256, 0, stream>>>(ej, ej16, nEJ);
  k_cvt_f16<<<(int)(nW  / 8 / 256), 256, 0, stream>>>(Wq, wq16, nW);
  k_cvt_f16<<<(int)(nW  / 8 / 256), 256, 0, stream>>>(Wk, wk16, nW);
  k_cvt_f16<<<(int)(nW  / 8 / 256), 256, 0, stream>>>(Wv, wv16, nW);
  k_cvt_f16<<<(int)(nW  / 8 / 256), 256, 0, stream>>>(Wc, wc16, nW);
  k_temb<<<B_, 256, 0, stream>>>(t, tW1, tb1, tW2, tb2, temb);

  dim3 g1(64, 8);                  // M=8192 / 128, C=1024 / 128
  k_qkv_gemm<<<g1, 256, 0, stream>>>(ej16, wq16, q16);
  k_qkv_gemm<<<g1, 256, 0, stream>>>(ej16, wk16, k16);
  k_qkv_gemm<<<g1, 256, 0, stream>>>(ej16, wv16, v16);

  dim3 ga(N_ / 128, H_, B_);       // (8, 16, 8)
  k_attn<<<ga, 256, 0, stream>>>(q16, k16, v16, oc16);

  k_mh_gemm<<<g1, 256, 0, stream>>>(oc16, wc16, bc, temb, mh16);

  dim3 gs(N_ / 128, N_ / 128, B_); // (8, 8, 8)
  k_score_conv<<<gs, 256, 0, stream>>>(mh16, ej16, xt, cw, cb, out);
}
